// SingleHeadAttention_30966714204282
// MI455X (gfx1250) — matrix operand order repaired and checked by reading, not compiled, not run on hardware
//
#include <hip/hip_runtime.h>

typedef __bf16 bf16;
typedef __bf16 bf16x8 __attribute__((ext_vector_type(8)));
typedef __bf16 v16bf  __attribute__((ext_vector_type(16)));
typedef float  v8f    __attribute__((ext_vector_type(8)));
typedef int    v4i    __attribute__((ext_vector_type(4)));

typedef __attribute__((address_space(1))) v4i* gbl_v4i_ptr;   // printed as 'int __vector(4) __device__ *'
typedef __attribute__((address_space(3))) v4i* lds_v4i_ptr;   // printed as 'int __vector(4) __shared__ *'

#define D_DIM 768
#define S_LEN 4096
#define B_SZ  4
#define M_TOT (B_SZ * S_LEN)   // 16384

#if __has_builtin(__builtin_amdgcn_global_load_async_to_lds_b128)
#define HAVE_ASYNC_LDS 1
#else
#define HAVE_ASYNC_LDS 0
#endif

// 16-byte global -> LDS copy. Prefer the CDNA5 async direct-to-LDS path
// (no VGPR round trip, tracked by ASYNCcnt); fall back to load+store.
__device__ __forceinline__ void copy16_g2s(bf16* ldst, const bf16* gsrc) {
#if HAVE_ASYNC_LDS
  __builtin_amdgcn_global_load_async_to_lds_b128(
      (gbl_v4i_ptr)gsrc, (lds_v4i_ptr)ldst, 0, 0);
#else
  *(bf16x8*)ldst = *(const bf16x8*)gsrc;
#endif
}

__device__ __forceinline__ void wait_lds_copies() {
#if HAVE_ASYNC_LDS
#if __has_builtin(__builtin_amdgcn_s_wait_asynccnt)
  __builtin_amdgcn_s_wait_asynccnt(0);
#else
  asm volatile("s_wait_asynccnt 0x0" ::: "memory");
#endif
#endif
}

// ---- WMMA fragment loaders (bf16, row-major K-contiguous source) ----
// A-matrix 16x32 (16-bit): lanes 0-15 row M=lane, K={k0..k0+7, k0+16..k0+23};
//                          lanes 16-31 row M=lane-16, K={k0+8..15, k0+24..31}.
__device__ __forceinline__ v16bf load_a_frag(const bf16* base, int ld, int k0) {
  int lane = threadIdx.x & 31;
  int idx  = lane & 15;
  int half = lane >> 4;
  const bf16* p = base + idx * ld + k0 + half * 8;
  bf16x8 lo = *(const bf16x8*)(p);
  bf16x8 hi = *(const bf16x8*)(p + 16);
  return __builtin_shufflevector(lo, hi, 0,1,2,3,4,5,6,7,8,9,10,11,12,13,14,15);
}
// B-matrix 32x16 (16-bit): lane = column N; lanes 0-15 hold K={k0..k0+7, k0+16..k0+23},
//                          lanes 16-31 hold K={k0+8..15, k0+24..31}.
__device__ __forceinline__ v16bf load_b_frag(const bf16* base, int ld, int k0) {
  int lane = threadIdx.x & 31;
  int idx  = lane & 15;
  int half = lane >> 4;
  const bf16* p = base + idx * ld + k0 + half * 8;
  bf16x8 lo = *(const bf16x8*)(p);
  bf16x8 hi = *(const bf16x8*)(p + 16);
  return __builtin_shufflevector(lo, hi, 0,1,2,3,4,5,6,7,8,9,10,11,12,13,14,15);
}

__device__ __forceinline__ v8f wmma_bf16(v16bf a, v16bf b, v8f c) {
  return __builtin_amdgcn_wmma_f32_16x16x32_bf16(false, a, false, b, (short)0, c, false, false);
}

// ---------------- conversion kernels ----------------
__global__ void cvt_x_kernel(const float* __restrict__ x, bf16* __restrict__ xb, int n) {
  int i = blockIdx.x * blockDim.x + threadIdx.x;
  if (i < n) xb[i] = (bf16)x[i];
}

// WT[n][k] = (bf16) W[k][n]  for Wq/Wk/Wv selected by blockIdx.y
__global__ void cvt_w_kernel(const float* __restrict__ Wq, const float* __restrict__ Wk,
                             const float* __restrict__ Wv,
                             bf16* __restrict__ WqT, bf16* __restrict__ WkT,
                             bf16* __restrict__ WvT) {
  int t = blockIdx.x * blockDim.x + threadIdx.x;           // 0 .. 768*768-1
  const float* W = (blockIdx.y == 0) ? Wq : (blockIdx.y == 1) ? Wk : Wv;
  bf16*       WT = (blockIdx.y == 0) ? WqT : (blockIdx.y == 1) ? WkT : WvT;
  int n = t / D_DIM, k = t % D_DIM;
  WT[t] = (bf16)W[k * D_DIM + n];
}

// ---------------- projection GEMM: out = X(16384x768) * W(768x768) + b ----------------
// 8 waves/WG: 4 M-subtiles x 2 N-groups(64). Each wave: 16x64 out (4 WMMA tiles).
// transposed==0 : out[m][n] bf16 row-major (for Q, K)
// transposed==1 : out stored as [b][n][s]   (for V, pre-transposed for P*V B-frags)
__global__ void proj_kernel(const bf16* __restrict__ Xb, const bf16* __restrict__ WT,
                            const float* __restrict__ bias, bf16* __restrict__ out,
                            int transposed) {
  int m0 = blockIdx.x * 64;
  int n0 = blockIdx.y * 128;
  int w    = threadIdx.x >> 5;
  int lane = threadIdx.x & 31;
  int idx  = lane & 15;
  int half = lane >> 4;
  int m_w = m0 + (w & 3) * 16;
  int n_w = n0 + (w >> 2) * 64;

  v8f c[4] = {};
  const bf16* arow = Xb + (size_t)m_w * D_DIM;
#pragma unroll 2
  for (int k0 = 0; k0 < D_DIM; k0 += 32) {
    v16bf a = load_a_frag(arow, D_DIM, k0);
#pragma unroll
    for (int t = 0; t < 4; ++t) {
      v16bf b = load_b_frag(WT + (size_t)(n_w + t * 16) * D_DIM, D_DIM, k0);
      c[t] = wmma_bf16(a, b, c[t]);
    }
  }
#pragma unroll
  for (int t = 0; t < 4; ++t) {
    int col = n_w + t * 16 + idx;
    float bv = bias[col];
#pragma unroll
    for (int g = 0; g < 8; ++g) {
      int row = m_w + g + 8 * half;
      float val = c[t][g] + bv;
      if (!transposed) {
        out[(size_t)row * D_DIM + col] = (bf16)val;
      } else {
        int bb = row >> 12;            // /4096
        int s  = row & (S_LEN - 1);
        out[((size_t)(bb * D_DIM + col)) * S_LEN + s] = (bf16)val;
      }
    }
  }
}

// ---------------- flash attention ----------------
// WG = 32 queries, 8 waves. KV tiles of 64 keys.
// Score phase: wave w -> (qi = w&1, kj = w>>1) 16x16 tile, K-loop over 768.
// PV phase:    wave w -> (qi = w&1, dg = w>>1) 16 q x 192 dims (12 accum tiles).
__global__ __launch_bounds__(256)
void attn_kernel(const bf16* __restrict__ Q, const bf16* __restrict__ K,
                 const bf16* __restrict__ Vt, float* __restrict__ out) {
  __shared__ __align__(16) bf16  Qs[32][D_DIM];    // 48 KB
  __shared__ __align__(16) bf16  Ks[64][D_DIM];    // 96 KB
  __shared__ __align__(16) bf16  Vts[D_DIM][64];   // 96 KB (d-major)
  __shared__ __align__(16) float Ss[32][64];       // 8 KB
  __shared__ __align__(16) bf16  Ps[32][64];       // 4 KB
  __shared__ float alpha_s[32];
  __shared__ float l_s[32];

  int b   = blockIdx.y;
  int q0  = blockIdx.x * 32;
  int tid = threadIdx.x;
  int w    = tid >> 5;
  int lane = tid & 31;
  int idx  = lane & 15;
  int half = lane >> 4;
  int qi = w & 1;
  int kj = w >> 1;     // score kv subtile
  int dg = w >> 1;     // PV d-group (192 dims)

  // load Q tile once (32 x 768), async direct-to-LDS
  const bf16* Qg = Q + ((size_t)(b * S_LEN + q0)) * D_DIM;
  for (int c = tid; c < 32 * (D_DIM / 8); c += 256) {
    int r = c / (D_DIM / 8), dc = c % (D_DIM / 8);
    copy16_g2s(&Qs[r][dc * 8], Qg + (size_t)r * D_DIM + dc * 8);
  }

  v8f o[12] = {};
  int   srow = tid >> 3;          // softmax: 8 threads per query row
  int   sj   = tid & 7;
  float m_prev = -1e30f;
  float l_run  = 0.0f;
  const float scale = 0.03608439182435161f;  // 1/sqrt(768)

  for (int kv0 = 0; kv0 < S_LEN; kv0 += 64) {
    __syncthreads();   // prev PV done (and Qs issue point on first iter)
    // load K tile 64x768
    const bf16* Kg = K + ((size_t)(b * S_LEN + kv0)) * D_DIM;
    for (int c = tid; c < 64 * (D_DIM / 8); c += 256) {
      int r = c / (D_DIM / 8), dc = c % (D_DIM / 8);
      copy16_g2s(&Ks[r][dc * 8], Kg + (size_t)r * D_DIM + dc * 8);
    }
    // load Vt tile 768x64 (global Vt is [b][d][s])
    const bf16* Vg = Vt + (size_t)b * D_DIM * S_LEN + kv0;
    for (int c = tid; c < D_DIM * 8; c += 256) {
      int d = c / 8, j = c % 8;
      copy16_g2s(&Vts[d][j * 8], Vg + (size_t)d * S_LEN + j * 8);
    }
    wait_lds_copies();   // ASYNCcnt -> 0 before the barrier
    __syncthreads();

    // scores: Q(16x768) x K^T(768x16)
    {
      v8f sc = {};
#pragma unroll 4
      for (int k0 = 0; k0 < D_DIM; k0 += 32) {
        v16bf a = load_a_frag(&Qs[qi * 16][0], D_DIM, k0);
        v16bf bb = load_b_frag(&Ks[kj * 16][0], D_DIM, k0);
        sc = wmma_bf16(a, bb, sc);
      }
      int col = kj * 16 + idx;
#pragma unroll
      for (int g = 0; g < 8; ++g)
        Ss[qi * 16 + g + 8 * half][col] = sc[g] * scale;
    }
    __syncthreads();

    // online softmax over this 64-wide slab
    {
      float v[8];
      float mloc = -1e30f;
#pragma unroll
      for (int e = 0; e < 8; ++e) { v[e] = Ss[srow][sj + e * 8]; mloc = fmaxf(mloc, v[e]); }
#pragma unroll
      for (int msk = 1; msk < 8; msk <<= 1) mloc = fmaxf(mloc, __shfl_xor(mloc, msk, 32));
      float m_new = fmaxf(m_prev, mloc);
      float alpha = __expf(m_prev - m_new);
      float ssum = 0.0f;
#pragma unroll
      for (int e = 0; e < 8; ++e) {
        float p = __expf(v[e] - m_new);
        ssum += p;
        Ps[srow][sj + e * 8] = (bf16)p;
      }
#pragma unroll
      for (int msk = 1; msk < 8; msk <<= 1) ssum += __shfl_xor(ssum, msk, 32);
      l_run = l_run * alpha + ssum;
      m_prev = m_new;
      if (sj == 0) { alpha_s[srow] = alpha; l_s[srow] = l_run; }
    }
    __syncthreads();

    // rescale accumulators and accumulate P(16x64) x V(64x192)
    {
      float al[8];
#pragma unroll
      for (int g = 0; g < 8; ++g) al[g] = alpha_s[qi * 16 + g + 8 * half];
#pragma unroll
      for (int t = 0; t < 12; ++t)
#pragma unroll
        for (int g = 0; g < 8; ++g) o[t][g] *= al[g];

#pragma unroll
      for (int k0 = 0; k0 < 64; k0 += 32) {
        v16bf a = load_a_frag(&Ps[qi * 16][0], 64, k0);
#pragma unroll
        for (int t = 0; t < 12; ++t) {
          v16bf bb = load_b_frag(&Vts[dg * 192 + t * 16][0], 64, k0);
          o[t] = wmma_bf16(a, bb, o[t]);
        }
      }
    }
  }
  __syncthreads();

  // normalize and write fp32 output
  float linv[8];
#pragma unroll
  for (int g = 0; g < 8; ++g) linv[g] = 1.0f / l_s[qi * 16 + g + 8 * half];
#pragma unroll
  for (int t = 0; t < 12; ++t) {
    int col = dg * 192 + t * 16 + idx;
#pragma unroll
    for (int g = 0; g < 8; ++g) {
      int row = q0 + qi * 16 + g + 8 * half;
      out[((size_t)(b * S_LEN + row)) * D_DIM + col] = o[t][g] * linv[g];
    }
  }
}

extern "C" void kernel_launch(void* const* d_in, const int* in_sizes, int n_in,
                              void* d_out, int out_size, void* d_ws, size_t ws_size,
                              hipStream_t stream) {
  const float* x  = (const float*)d_in[0];
  const float* Wq = (const float*)d_in[1];
  const float* bq = (const float*)d_in[2];
  const float* Wk = (const float*)d_in[3];
  const float* bk = (const float*)d_in[4];
  const float* Wv = (const float*)d_in[5];
  const float* bv = (const float*)d_in[6];
  float* out = (float*)d_out;

  // workspace layout (bytes)
  char* ws = (char*)d_ws;
  bf16* xb  = (bf16*)(ws + 0);           // 16384*768*2 = 25,165,824
  bf16* WqT = (bf16*)(ws + 25165824);    // 768*768*2   =  1,179,648
  bf16* WkT = (bf16*)(ws + 26345472);
  bf16* WvT = (bf16*)(ws + 27525120);
  bf16* qb  = (bf16*)(ws + 28704768);    // 25,165,824
  bf16* kb  = (bf16*)(ws + 53870592);    // 25,165,824
  bf16* vtb = (bf16*)(ws + 79036416);    // 25,165,824  (stored [b][d][s])

  int nx = M_TOT * D_DIM;
  cvt_x_kernel<<<nx / 256, 256, 0, stream>>>(x, xb, nx);
  cvt_w_kernel<<<dim3((D_DIM * D_DIM) / 256, 3), 256, 0, stream>>>(Wq, Wk, Wv, WqT, WkT, WvT);

  dim3 pg(M_TOT / 64, D_DIM / 128);
  proj_kernel<<<pg, 256, 0, stream>>>(xb, WqT, bq, qb, 0);
  proj_kernel<<<pg, 256, 0, stream>>>(xb, WkT, bk, kb, 0);
  proj_kernel<<<pg, 256, 0, stream>>>(xb, WvT, bv, vtb, 1);

  attn_kernel<<<dim3(S_LEN / 32, B_SZ), 256, 0, stream>>>(qb, kb, vtb, out);
}
